// Attention_34187939676584
// MI455X (gfx1250) — matrix operand order repaired and checked by reading, not compiled, not run on hardware
//
#include <hip/hip_runtime.h>
#include <hip/hip_bf16.h>
#include <stdint.h>

// ---------------- types ----------------
typedef __attribute__((ext_vector_type(16))) _Float16 v16h;
typedef __attribute__((ext_vector_type(8)))  _Float16 h8;
typedef __attribute__((ext_vector_type(8)))  float    v8f;

#define HEADS  8
#define DHEAD  64
#define CIN    256
#define NPIX   4096      // 64*64 per image
#define NCOL   8192      // 2 * 4096 pixels total
#define KVPAD  4128      // 4096 computed + 4 mem tokens + 28 pad (multiple of 32)
#define KVREAL 4100

// D = A(16x32 f16) * B(32x16 f16) + C(16x16 f32)
__device__ __forceinline__ v8f wmma16(v16h a, v16h b, v8f c) {
  return __builtin_amdgcn_wmma_f32_16x16x32_f16(false, a, false, b, (short)0, c,
                                                false, false);
}

// A-fragment (16x32 f16): lane (l&15)=row M; lane half (l>>4) selects K subsets.
// base must point at row_start + k0 + (l>>4)*16; elements 0..15 contiguous.
__device__ __forceinline__ v16h load_a_frag(const _Float16* base) {
  return *(const v16h*)(base);
}

// CDNA5 async copy: LDS[ldsOff] = MEM[gaddr], 16 bytes per lane, ASYNCcnt-tracked
__device__ __forceinline__ void async_b128_to_lds(unsigned ldsOff, const void* gaddr) {
  asm volatile("global_load_async_to_lds_b128 %0, %1, off"
               :: "v"(ldsOff), "v"((uint64_t)(uintptr_t)gaddr) : "memory");
}
__device__ __forceinline__ void wait_asynccnt0() {
  asm volatile("s_wait_asynccnt 0x0" ::: "memory");
}

// ---------------- P0: weight conversion + mem_kv/pad init ----------------
__global__ void k_prep(const float* __restrict__ wqkv, const float* __restrict__ wout,
                       const float* __restrict__ memkv,
                       _Float16* __restrict__ wq16, _Float16* __restrict__ wo16,
                       _Float16* __restrict__ Kb, _Float16* __restrict__ Vb) {
  int idx = blockIdx.x * 256 + threadIdx.x;
  if (idx < 1536 * 256) wq16[idx] = (_Float16)wqkv[idx];
  if (idx < 256 * 512)  wo16[idx] = (_Float16)wout[idx];
  if (idx < 2 * HEADS * 32 * DHEAD) {          // rows 4096..4127 of K/V per (b,head)
    int d = idx & 63; int t = idx >> 6;
    int row = t & 31; t >>= 5;
    int head = t & 7; int b = t >> 3;
    float vk = 0.f, vv = 0.f;
    if (row < 4) {                              // mem tokens appended after real keys
      vk = memkv[(head * 4 + row) * 64 + d];                   // mem_kv[0]
      vv = memkv[HEADS * 4 * 64 + (head * 4 + row) * 64 + d];  // mem_kv[1]
    }
    size_t o = ((size_t)(b * HEADS + head) * KVPAD + 4096 + row) * 64 + d;
    Kb[o] = (_Float16)vk;
    Vb[o] = (_Float16)vv;
  }
}

// ---------------- P1: RMSNorm -> xn^T (f16, [pixel][channel]) ----------------
__global__ void k_norm(const float* __restrict__ x, const float* __restrict__ gamma,
                       _Float16* __restrict__ xnT) {
  int col = blockIdx.x;                 // 0..8191
  int b = col >> 12, p = col & 4095;
  int c = threadIdx.x;                  // 0..255
  float v = x[((size_t)b * CIN + c) * NPIX + p];
  __shared__ float red[256];
  red[c] = v * v;
  __syncthreads();
#pragma unroll
  for (int s = 128; s > 0; s >>= 1) {
    if (c < s) red[c] += red[c + s];
    __syncthreads();
  }
  float norm = fmaxf(sqrtf(red[0]), 1e-12f);
  float scale = (gamma[c] + 1.0f) * 16.0f / norm;   // sqrt(256) = 16
  xnT[(size_t)col * CIN + c] = (_Float16)(v * scale);
}

// ---------------- P2: QKV GEMM [1536x256]x[256x8192], wave tile 64x64 --------
__global__ void __launch_bounds__(32, 1)
k_qkv(const _Float16* __restrict__ w16, const _Float16* __restrict__ xnT,
      _Float16* __restrict__ Qb, _Float16* __restrict__ Kb,
      _Float16* __restrict__ Vb) {
  const int lane = threadIdx.x & 31;
  const int lo = lane & 15, hi = lane >> 4;
  const int m0 = blockIdx.x * 64, n0 = blockIdx.y * 64;

  // lane-constant base pointers; k0 folds into load immediates after full unroll
  const _Float16* aRow[4];
  const _Float16* bCol[4];
#pragma unroll
  for (int mt = 0; mt < 4; ++mt)
    aRow[mt] = w16 + (m0 + mt * 16 + lo) * CIN + hi * 16;
#pragma unroll
  for (int nt = 0; nt < 4; ++nt)
    bCol[nt] = xnT + (n0 + nt * 16 + lo) * CIN + hi * 16;

  v8f acc[4][4];
#pragma unroll
  for (int i = 0; i < 4; ++i)
#pragma unroll
    for (int j = 0; j < 4; ++j)
#pragma unroll
      for (int r = 0; r < 8; ++r) acc[i][j][r] = 0.f;

#pragma unroll
  for (int k0 = 0; k0 < CIN; k0 += 32) {
    v16h a[4], bf[4];
#pragma unroll
    for (int mt = 0; mt < 4; ++mt) a[mt] = load_a_frag(aRow[mt] + k0);
#pragma unroll
    for (int nt = 0; nt < 4; ++nt) bf[nt] = *(const v16h*)(bCol[nt] + k0);
#pragma unroll
    for (int mt = 0; mt < 4; ++mt)
#pragma unroll
      for (int nt = 0; nt < 4; ++nt)
        acc[mt][nt] = wmma16(a[mt], bf[nt], acc[mt][nt]);
  }

  // block-uniform q/k/v routing (64-row tile lies inside one 512-row segment)
  const int seg = m0 >> 9;                          // 0=q, 1=k, 2=v
  _Float16* dst = (seg == 0) ? Qb : ((seg == 1) ? Kb : Vb);
  const int rows = (seg == 0) ? NPIX : KVPAD;
  const float qs = (seg == 0) ? 0.125f : 1.0f;      // DIM_HEAD ** -0.5 folded into q
  const int obase = m0 - seg * 512;
#pragma unroll
  for (int mt = 0; mt < 4; ++mt)
#pragma unroll
    for (int r = 0; r < 8; ++r) {
      int o2 = obase + mt * 16 + 8 * hi + r;
      int head = o2 >> 6, d = o2 & 63;
#pragma unroll
      for (int nt = 0; nt < 4; ++nt) {
        int col = n0 + nt * 16 + lo;
        int b = col >> 12, n = col & 4095;
        dst[((size_t)(b * HEADS + head) * rows + n) * 64 + d] =
            (_Float16)(acc[mt][nt][r] * qs);
      }
    }
}

// ---------------- P3: flash attention, 4 waves/block, 32 queries/wave --------
__global__ void __launch_bounds__(128, 1)
k_attn(const _Float16* __restrict__ Qb, const _Float16* __restrict__ Kb,
       const _Float16* __restrict__ Vb, _Float16* __restrict__ Ob) {
  const int tid = threadIdx.x;
  const int w = tid >> 5, lane = tid & 31;
  const int lo = lane & 15, hi = lane >> 4;
  const int bh = blockIdx.x;                 // 0..15
  const int b = bh >> 3, head = bh & 7;
  const int q0 = blockIdx.y * 128 + w * 32;

  const _Float16* Qh = Qb + (size_t)bh * NPIX * 64;
  const _Float16* Vh = Vb + (size_t)bh * KVPAD * 64;

  __shared__ alignas(32) _Float16 kT[32 * 64];       // [key][d]  (async-filled)
  __shared__ alignas(32) _Float16 vT[64 * 32];       // [d][key]  (transposed)
  __shared__ alignas(32) _Float16 pB[4][32 * 32];    // per-wave P staging [q][key]

  // ---- lane-constant LDS fragment pointers (hoisted out of the 129-tile loop)
  const _Float16* kbPtr[2][2];                       // [nt][ks]
#pragma unroll
  for (int nt = 0; nt < 2; ++nt)
#pragma unroll
    for (int ks = 0; ks < 2; ++ks)
      kbPtr[nt][ks] = kT + (nt * 16 + lo) * 64 + ks * 32 + hi * 16;
  const _Float16* vbPtr[4];
#pragma unroll
  for (int nd = 0; nd < 4; ++nd)
    vbPtr[nd] = vT + (nd * 16 + lo) * 32 + hi * 16;
  const _Float16* paPtr[2];
#pragma unroll
  for (int mt = 0; mt < 2; ++mt)
    paPtr[mt] = &pB[w][(mt * 16 + lo) * 32 + hi * 16];
  _Float16* pSt[2];
#pragma unroll
  for (int mt = 0; mt < 2; ++mt)
    pSt[mt] = &pB[w][(mt * 16 + 8 * hi) * 32 + lo];  // + r*32 per row, +16 for nt=1

  // ---- cooperative-load coordinates / pointers
  const int row = tid >> 2, seg = tid & 3;           // 32 rows x 4 segs of 16 halves
  const _Float16* vLd = Vh + row * 64 + seg * 16;    // advances 2048/iter
  _Float16* vSt = vT + seg * 16 * 32 + row;          // lane-constant
  const unsigned ldsK = (unsigned)(uintptr_t)&kT[0]; // LDS byte offset of kT
  const unsigned ldsK0 = ldsK + (unsigned)tid * 16;
  const _Float16* kLd = Kb + (size_t)bh * KVPAD * 64 + tid * 8;  // 16B chunk / lane

  // ---- Q fragments: [mt][kstep over d]
  v16h qa[2][2];
#pragma unroll
  for (int mt = 0; mt < 2; ++mt)
#pragma unroll
    for (int ks = 0; ks < 2; ++ks)
      qa[mt][ks] = load_a_frag(Qh + (q0 + mt * 16 + lo) * 64 + ks * 32 + hi * 16);

  v8f oAcc[2][4];
#pragma unroll
  for (int i = 0; i < 2; ++i)
#pragma unroll
    for (int j = 0; j < 4; ++j)
#pragma unroll
      for (int r = 0; r < 8; ++r) oAcc[i][j][r] = 0.f;
  float rowMax[2][8], rowSum[2][8];
#pragma unroll
  for (int i = 0; i < 2; ++i)
#pragma unroll
    for (int r = 0; r < 8; ++r) { rowMax[i][r] = -1e30f; rowSum[i][r] = 0.f; }

  for (int jt = 0; jt < KVPAD / 32; ++jt) {  // 129 tiles of 32 keys
    __syncthreads();                          // previous tile fully consumed
    // K tile: 4KB via 2x async b128 per lane, straight into LDS (ASYNCcnt)
    async_b128_to_lds(ldsK0, kLd);
    async_b128_to_lds(ldsK0 + 2048, kLd + 1024);
    // V tile: through VGPRs (needs [d][key] transpose)
    {
      h8 va = *(const h8*)(vLd), vb2 = *(const h8*)(vLd + 8);
#pragma unroll
      for (int j = 0; j < 8; ++j) {
        vSt[j * 32] = va[j];
        vSt[(8 + j) * 32] = vb2[j];
      }
    }
    if (jt + 1 < KVPAD / 32)                  // prefetch next V tile
      __builtin_prefetch(vLd + 2048, 0, 1);
    kLd += 2048;
    vLd += 2048;
    wait_asynccnt0();
    __syncthreads();

    // S = Q * K^T  (32x32 tile, fp32 accum)
    v8f s[2][2];
#pragma unroll
    for (int i = 0; i < 2; ++i)
#pragma unroll
      for (int j = 0; j < 2; ++j)
#pragma unroll
        for (int r = 0; r < 8; ++r) s[i][j][r] = 0.f;
#pragma unroll
    for (int ks = 0; ks < 2; ++ks) {
      v16h kb0 = *(const v16h*)(kbPtr[0][ks]);
      v16h kb1 = *(const v16h*)(kbPtr[1][ks]);
#pragma unroll
      for (int mt = 0; mt < 2; ++mt) {
        s[mt][0] = wmma16(qa[mt][ks], kb0, s[mt][0]);
        s[mt][1] = wmma16(qa[mt][ks], kb1, s[mt][1]);
      }
    }

    if (jt == KVPAD / 32 - 1) {               // mask padding keys (j >= 4100)
#pragma unroll
      for (int nt = 0; nt < 2; ++nt)
        if (jt * 32 + nt * 16 + lo >= KVREAL) {
#pragma unroll
          for (int mt = 0; mt < 2; ++mt)
#pragma unroll
            for (int r = 0; r < 8; ++r) s[mt][nt][r] = -1e30f;
        }
    }

    // online softmax: row reductions across 16-lane groups (C-layout rows)
#pragma unroll
    for (int mt = 0; mt < 2; ++mt)
#pragma unroll
      for (int r = 0; r < 8; ++r) {
        float tmax = fmaxf(s[mt][0][r], s[mt][1][r]);
        tmax = fmaxf(tmax, __shfl_xor(tmax, 1, 32));
        tmax = fmaxf(tmax, __shfl_xor(tmax, 2, 32));
        tmax = fmaxf(tmax, __shfl_xor(tmax, 4, 32));
        tmax = fmaxf(tmax, __shfl_xor(tmax, 8, 32));
        float mold = rowMax[mt][r];
        float mnew = fmaxf(mold, tmax);
        float alpha = __expf(mold - mnew);
        rowMax[mt][r] = mnew;
        float p0 = __expf(s[mt][0][r] - mnew);
        float p1 = __expf(s[mt][1][r] - mnew);
        float ps = p0 + p1;
        ps += __shfl_xor(ps, 1, 32);
        ps += __shfl_xor(ps, 2, 32);
        ps += __shfl_xor(ps, 4, 32);
        ps += __shfl_xor(ps, 8, 32);
        rowSum[mt][r] = rowSum[mt][r] * alpha + ps;
#pragma unroll
        for (int nd = 0; nd < 4; ++nd) oAcc[mt][nd][r] *= alpha;
        pSt[mt][r * 32] = (_Float16)p0;
        pSt[mt][r * 32 + 16] = (_Float16)p1;
      }
    __syncthreads();

    // O += P * V
    v16h pa0 = load_a_frag(paPtr[0]);
    v16h pa1 = load_a_frag(paPtr[1]);
#pragma unroll
    for (int nd = 0; nd < 4; ++nd) {
      v16h vb = *(const v16h*)(vbPtr[nd]);
      oAcc[0][nd] = wmma16(pa0, vb, oAcc[0][nd]);
      oAcc[1][nd] = wmma16(pa1, vb, oAcc[1][nd]);
    }
  }

  // epilogue: normalize and store O as [pixel][head*64+d] f16
#pragma unroll
  for (int mt = 0; mt < 2; ++mt)
#pragma unroll
    for (int r = 0; r < 8; ++r) {
      float inv = 1.f / rowSum[mt][r];
      int n = q0 + mt * 16 + 8 * hi + r;
#pragma unroll
      for (int nd = 0; nd < 4; ++nd) {
        int d = nd * 16 + lo;
        Ob[((size_t)(b * NPIX + n)) * 512 + head * 64 + d] =
            (_Float16)(oAcc[mt][nd][r] * inv);
      }
    }
}

// ---------------- P4: output GEMM [256x512]x[512x8192] -> fp32 out ----------
__global__ void __launch_bounds__(32, 1)
k_out(const _Float16* __restrict__ wo16, const _Float16* __restrict__ Ob,
      float* __restrict__ out) {
  const int lane = threadIdx.x & 31;
  const int lo = lane & 15, hi = lane >> 4;
  const int m0 = blockIdx.x * 64, n0 = blockIdx.y * 64;

  const _Float16* aRow[4];
  const _Float16* bCol[4];
#pragma unroll
  for (int mt = 0; mt < 4; ++mt)
    aRow[mt] = wo16 + (m0 + mt * 16 + lo) * 512 + hi * 16;
#pragma unroll
  for (int nt = 0; nt < 4; ++nt)
    bCol[nt] = Ob + (n0 + nt * 16 + lo) * 512 + hi * 16;

  v8f acc[4][4];
#pragma unroll
  for (int i = 0; i < 4; ++i)
#pragma unroll
    for (int j = 0; j < 4; ++j)
#pragma unroll
      for (int r = 0; r < 8; ++r) acc[i][j][r] = 0.f;

#pragma unroll
  for (int k0 = 0; k0 < 512; k0 += 32) {
    v16h a[4], bf[4];
#pragma unroll
    for (int mt = 0; mt < 4; ++mt) a[mt] = load_a_frag(aRow[mt] + k0);
#pragma unroll
    for (int nt = 0; nt < 4; ++nt) bf[nt] = *(const v16h*)(bCol[nt] + k0);
#pragma unroll
    for (int mt = 0; mt < 4; ++mt)
#pragma unroll
      for (int nt = 0; nt < 4; ++nt)
        acc[mt][nt] = wmma16(a[mt], bf[nt], acc[mt][nt]);
  }

#pragma unroll
  for (int mt = 0; mt < 4; ++mt)
#pragma unroll
    for (int nt = 0; nt < 4; ++nt)
#pragma unroll
      for (int r = 0; r < 8; ++r) {
        int o = m0 + mt * 16 + 8 * hi + r;
        int col = n0 + nt * 16 + lo;
        int b = col >> 12, p = col & 4095;
        out[((size_t)(b * CIN + o)) * NPIX + p] = acc[mt][nt][r];
      }
}

// ---------------- launch ----------------
extern "C" void kernel_launch(void* const* d_in, const int* in_sizes, int n_in,
                              void* d_out, int out_size, void* d_ws, size_t ws_size,
                              hipStream_t stream) {
  (void)in_sizes; (void)n_in; (void)out_size; (void)ws_size;
  const float* x     = (const float*)d_in[0];
  const float* gamma = (const float*)d_in[1];
  const float* memkv = (const float*)d_in[2];
  const float* wqkv  = (const float*)d_in[3];
  const float* wout  = (const float*)d_in[4];
  float* out = (float*)d_out;

  char* ws = (char*)d_ws;
  size_t off = 0;
  auto carve = [&](size_t bytes) -> char* {
    char* p = ws + off;
    off += (bytes + 255) & ~(size_t)255;
    return p;
  };
  _Float16* xnT  = (_Float16*)carve((size_t)NCOL * CIN * 2);              // 4 MB
  _Float16* wq16 = (_Float16*)carve((size_t)1536 * 256 * 2);              // 0.75 MB
  _Float16* wo16 = (_Float16*)carve((size_t)256 * 512 * 2);               // 0.25 MB
  _Float16* Qb   = (_Float16*)carve((size_t)2 * HEADS * NPIX * 64 * 2);   // 8 MB
  _Float16* Kb   = (_Float16*)carve((size_t)2 * HEADS * KVPAD * 64 * 2);  // 8.06 MB
  _Float16* Vb   = (_Float16*)carve((size_t)2 * HEADS * KVPAD * 64 * 2);  // 8.06 MB
  _Float16* Ob   = (_Float16*)carve((size_t)NCOL * 512 * 2);              // 8 MB

  k_prep<<<1536, 256, 0, stream>>>(wqkv, wout, memkv, wq16, wo16, Kb, Vb);
  k_norm<<<NCOL, 256, 0, stream>>>(x, gamma, xnT);
  k_qkv<<<dim3(1536 / 64, NCOL / 64), 32, 0, stream>>>(wq16, xnT, Qb, Kb, Vb);
  k_attn<<<dim3(2 * HEADS, NPIX / 128), 128, 0, stream>>>(Qb, Kb, Vb, Ob);
  k_out<<<dim3(256 / 64, NCOL / 64), 32, 0, stream>>>(wo16, Ob, out);
}
